// PWC3d_Lite_50981261803867
// MI455X (gfx1250) — compile-verified, matrix operand order checked by reading, not run on hardware
//
#include <hip/hip_runtime.h>
#include <hip/hip_bf16.h>
#include <cstdint>
#include <cstddef>

// ---------------------------------------------------------------------------
// Types
// ---------------------------------------------------------------------------
typedef __attribute__((ext_vector_type(16))) _Float16     v16h;
typedef __attribute__((ext_vector_type(8)))  float        v8f;
typedef __attribute__((ext_vector_type(4)))  unsigned int u32x4;
typedef __attribute__((ext_vector_type(8)))  int          i32x8;
typedef __attribute__((ext_vector_type(4)))  int          i32x4;

#define DEV_LRELU(v) ((v) > 0.f ? (v) : 0.1f * (v))

// ---------------------------------------------------------------------------
// Implicit-GEMM conv3d with WMMA f32_16x16x32_f16, channel-last activations.
//   Activations: f16, [voxel, Cstride]  (channel-last; Cstride multiple of 8)
//   GEMM: M = out voxels, N = Cout, K = KD^3 * CinP  (cin fastest, padded to 8)
//   Weights: f16, [Cout, K] permuted so k = spat*CinP + cin  (zero-padded)
// A-tile: each staging thread -> ONE global_load_b128 (8 contiguous channels
// of one voxel) + one ds_store_b128.  B-tile: Tensor Data Mover 2D tile with
// LDS row padding.  Per-wave 16x16 tile, 8 waves = 64(M) x 32(N) per block.
// ---------------------------------------------------------------------------
struct ConvArgs {
  const _Float16* __restrict__ in;
  const _Float16* __restrict__ wt;
  const float*    __restrict__ bias;
  _Float16* __restrict__ out16;   // channel-last, may be null
  float*    __restrict__ out32;   // channel-major [Cout][M], may be null
  int inStrideC;                  // halves per voxel in input buffer
  int outStrideC;                 // halves per voxel in f16 output buffer
  int Cin, CinP, Cout, K, M;
  int d;      // output cube dim (power of 2)
  int zb;     // log2(d)
  int inD;    // input cube dim
  int stride, dil, pad, relu;
};

template <int KD>
__global__ __launch_bounds__(256) void conv3d_wmma(ConvArgs a) {
  constexpr int K3 = KD * KD * KD;
  // A: 64 rows x 32 halves, padded to 40-half (80B) stride. B: 32 x 32 -> 40.
  __shared__ __align__(16) _Float16 As[64 * 40];
  __shared__ __align__(16) _Float16 Bs[32 * 40];

  const int tid  = threadIdx.x;
  const int lane = tid & 31;
  const int wv   = tid >> 5;
  const int hl   = lane >> 4;   // which half of the wave
  const int l15  = lane & 15;

  const int mBlock = blockIdx.x * 64;
  const int nBlock = blockIdx.y * 32;
  const int mBase  = mBlock + (wv & 3) * 16;
  const int nBase  = nBlock + (wv >> 2) * 16;

  // Staging: thread fills one aligned run of 8 halves of As.
  const int aRow   = tid >> 2;        // 0..63
  const int aCol   = (tid & 3) * 8;   // 0,8,16,24
  const int mStage = mBlock + aRow;
  const int mask   = a.d - 1;
  int ox = 0, oy = 0, oz = 0;
  const bool mOK = mStage < a.M;
  if (mOK) {
    oz = mStage & mask;
    oy = (mStage >> a.zb) & mask;
    ox = mStage >> (2 * a.zb);
  }

  // Incremental K decomposition: k = spat*CinP + cin, cin multiple of 8,
  // CinP multiple of 8 => an 8-run never straddles a voxel.
  int spat = 0, cin = aCol;
  while (cin >= a.CinP) { cin -= a.CinP; ++spat; }

  v8f acc = {};

  for (int kk0 = 0; kk0 < a.K; kk0 += 32) {
    // ---- B tile via Tensor Data Mover (one wave issues the DMA) ----------
    if (wv == 0) {
      const _Float16* gsrc = a.wt + (size_t)nBlock * a.K + kk0;
      unsigned long long ga = (unsigned long long)(size_t)gsrc;
      unsigned ldsOff = (unsigned)(size_t)(&Bs[0]);  // low 32 bits == LDS offset
      unsigned td0 = (unsigned)(a.K - kk0);          // remaining K (zero-fill past)
      unsigned td1 = (unsigned)(a.Cout - nBlock);    // remaining rows
      u32x4 g0;
      g0[0] = 1u;                                    // count=1, user descriptor
      g0[1] = ldsOff;                                // lds_addr
      g0[2] = (unsigned)(ga & 0xFFFFFFFFu);          // global_addr lo
      g0[3] = (unsigned)((ga >> 32) & 0x1FFFFFFu) | (2u << 30);  // hi | type=2
      i32x8 g1;
      // data_size=1(2B)<<16 | pad_enable<<20 | pad_interval=3(16dw)<<22
      // | pad_amount=3(4dw)<<25  => LDS row stride 64B + 16B pad = 80B
      g1[0] = (1 << 16) | (1 << 20) | (3 << 22) | (3 << 25);
      g1[1] = (int)((td0 & 0xFFFFu) << 16);
      g1[2] = (int)((td0 >> 16) | ((td1 & 0xFFFFu) << 16));
      g1[3] = (int)((td1 >> 16) | (32u << 16));      // tile_dim0 = 32 (K)
      g1[4] = (int)(32u);                            // tile_dim1 = 32 (N rows)
      unsigned long long st0 = (unsigned long long)a.K;  // row stride (elems)
      g1[5] = (int)(st0 & 0xFFFFFFFFu);
      g1[6] = (int)((st0 >> 32) & 0xFFFFu);
      g1[7] = 0;
      i32x4 z4 = {0, 0, 0, 0};
      i32x8 z8 = {0, 0, 0, 0, 0, 0, 0, 0};
      __builtin_amdgcn_tensor_load_to_lds(g0, g1, z4, z4, z8, 0);
      __builtin_amdgcn_s_wait_tensorcnt(0);
    }

    // ---- A tile: one predicated b128 load per thread -> LDS --------------
    uint4 v4 = {0u, 0u, 0u, 0u};
    if (mOK && spat < K3) {
      const int kx   = spat / (KD * KD);
      const int rem2 = spat - kx * KD * KD;
      const int ky   = rem2 / KD;
      const int kz   = rem2 - ky * KD;
      const int ix = ox * a.stride + a.dil * kx - a.pad;
      const int iy = oy * a.stride + a.dil * ky - a.pad;
      const int iz = oz * a.stride + a.dil * kz - a.pad;
      if ((unsigned)ix < (unsigned)a.inD && (unsigned)iy < (unsigned)a.inD &&
          (unsigned)iz < (unsigned)a.inD) {
        const int voxel = (ix * a.inD + iy) * a.inD + iz;
        const _Float16* p = a.in + (size_t)voxel * a.inStrideC + cin;
        v4 = *(const uint4*)p;
        __builtin_prefetch(p + a.inStrideC, 0, 1);  // next spatial tap
      }
    }
    *(uint4*)((char*)As + aRow * 80 + aCol * 2) = v4;

    __syncthreads();

    // ---- WMMA: build per-lane fragments from LDS (two b128 runs each) ----
    // 16-bit A 16x32 layout: lane<16 holds K {0..7,16..23}; lane>=16 holds
    // K {8..15,24..31}; run0 byte off = 16*hl, run1 = 32 + 16*hl.
    union { v16h h; uint4 u[2]; } av, bv;
    {
      const char* Ab = (const char*)As + ((wv & 3) * 16 + l15) * 80;
      av.u[0] = *(const uint4*)(Ab + 16 * hl);
      av.u[1] = *(const uint4*)(Ab + 32 + 16 * hl);
      const char* Bb = (const char*)Bs + ((wv >> 2) * 16 + l15) * 80;
      bv.u[0] = *(const uint4*)(Bb + 16 * hl);
      bv.u[1] = *(const uint4*)(Bb + 32 + 16 * hl);
    }
    acc = __builtin_amdgcn_wmma_f32_16x16x32_f16(
        false, av.h, false, bv.h, (short)0, acc, false, false);

    __syncthreads();

    // advance K decomposition by one 32-chunk
    cin += 32;
    while (cin >= a.CinP) { cin -= a.CinP; ++spat; }
  }

  // ---- Epilogue: bias + leaky ReLU; C/D layout: VGPR r -> M=r+8*hl, N=l15.
#pragma unroll
  for (int r = 0; r < 8; ++r) {
    const int m = mBase + r + 8 * hl;
    const int n = nBase + l15;
    if (m < a.M && n < a.Cout) {
      float v = acc[r] + a.bias[n];
      if (a.relu) v = DEV_LRELU(v);
      if (a.out16) a.out16[(size_t)m * a.outStrideC + n] = (_Float16)v;
      if (a.out32) a.out32[(size_t)n * a.M + m] = v;
    }
  }
}

// ---------------------------------------------------------------------------
// Weight permutation: OIDHW f32 -> [Cout, K] f16 with k = spat*CinP + cin,
// zero-padded for cin >= Cin (covers activation channel padding for free).
// ---------------------------------------------------------------------------
__global__ void permw_kernel(const float* __restrict__ w, _Float16* __restrict__ out,
                             int Cout, int Cin, int CinP, int K3, int K) {
  const int idx = blockIdx.x * blockDim.x + threadIdx.x;
  if (idx >= Cout * K) return;
  const int n = idx / K;
  const int r = idx - n * K;
  const int spat = r / CinP;
  const int cin  = r - spat * CinP;
  float v = 0.f;
  if (cin < Cin) v = w[((size_t)n * Cin + cin) * K3 + spat];
  out[idx] = (_Float16)v;
}

// ---------------------------------------------------------------------------
// Elementwise / layout helper kernels
// ---------------------------------------------------------------------------
// f32 [v] -> f16 channel-last [v,8] (channel 0 = value, 1..7 zero).
__global__ void cast_in_kernel(const float* __restrict__ in,
                               _Float16* __restrict__ out, int n) {
  const int i = blockIdx.x * blockDim.x + threadIdx.x;
  if (i >= n) return;
  union { _Float16 h[8]; uint4 u; } pk;
  pk.h[0] = (_Float16)in[i];
#pragma unroll
  for (int j = 1; j < 8; ++j) pk.h[j] = (_Float16)0.f;
  *(uint4*)(out + (size_t)i * 8) = pk.u;
}

// flow f32 channel-major [3][M] -> f16 channel-last dst[v*stride + ch].
__global__ void cast_flowcl_kernel(const float* __restrict__ flow,
                                   _Float16* __restrict__ dst, int M, int strideC) {
  const int i = blockIdx.x * blockDim.x + threadIdx.x;
  if (i >= 3 * M) return;
  const int ch = i / M;
  const int v  = i - ch * M;
  dst[(size_t)v * strideC + ch] = (_Float16)flow[i];
}

// strided channel-last copy: dst[v*ds + c] = src[v*ss + c], c < nch.
__global__ void copy_chcl_kernel(const _Float16* __restrict__ src,
                                 _Float16* __restrict__ dst,
                                 int M, int nch, int ss, int ds) {
  const int i = blockIdx.x * blockDim.x + threadIdx.x;
  if (i >= M * nch) return;
  const int v = i / nch;
  const int c = i - v * nch;
  dst[(size_t)v * ds + c] = src[(size_t)v * ss + c];
}

__global__ void fill_f32_kernel(float* __restrict__ p, float v, int n) {
  const int i = blockIdx.x * blockDim.x + threadIdx.x;
  if (i < n) p[i] = v;
}

__global__ void add_f32_kernel(float* __restrict__ a, const float* __restrict__ b, int n) {
  const int i = blockIdx.x * blockDim.x + threadIdx.x;
  if (i < n) a[i] += b[i];
}

// Correlation (channel-last): out[v*768 + disp] = lrelu(mean_c x1[v,c]*wp[sv,c])
// with zero outside the volume.  grid.y = 729 displacements.
__global__ void corr_kernel(const _Float16* __restrict__ x1f,
                            const _Float16* __restrict__ wp,
                            _Float16* __restrict__ out, int C, int Cs, int D,
                            int outStride) {
  const int XYZ = D * D * D;
  const int v = blockIdx.x * blockDim.x + threadIdx.x;
  const int disp = blockIdx.y;
  if (v >= XYZ) return;
  const int z = v % D, y = (v / D) % D, x = v / (D * D);
  const int dz = disp % 9 - 4, dy = (disp / 9) % 9 - 4, dx = disp / 81 - 4;
  const int sx = x + dx, sy = y + dy, sz = z + dz;
  float acc = 0.f;
  if ((unsigned)sx < (unsigned)D && (unsigned)sy < (unsigned)D &&
      (unsigned)sz < (unsigned)D) {
    const int sv = (sx * D + sy) * D + sz;
    const _Float16* p1 = x1f + (size_t)v * Cs;
    const _Float16* p2 = wp + (size_t)sv * Cs;
    float a0 = 0.f, a1 = 0.f;
    for (int c = 0; c < C; c += 2) {       // contiguous channel reads
      a0 += (float)p1[c] * (float)p2[c];
      a1 += (float)p1[c + 1] * (float)p2[c + 1];
    }
    acc = (a0 + a1) * (1.f / (float)C);
  }
  acc = DEV_LRELU(acc);
  out[(size_t)v * outStride + disp] = (_Float16)acc;
}

// Trilinear backward warp with border clamp, channel-last features.
__global__ void warp_kernel(const _Float16* __restrict__ src,
                            const float* __restrict__ flow,
                            _Float16* __restrict__ dst, int C, int D) {
  const int XYZ = D * D * D;
  const int v = blockIdx.x * blockDim.x + threadIdx.x;
  if (v >= XYZ) return;
  const int z = v % D, y = (v / D) % D, x = v / (D * D);
  const float fm = (float)(D - 1);
  float px = (float)x + flow[v];            px = px < 0.f ? 0.f : (px > fm ? fm : px);
  float py = (float)y + flow[XYZ + v];      py = py < 0.f ? 0.f : (py > fm ? fm : py);
  float pz = (float)z + flow[2 * XYZ + v];  pz = pz < 0.f ? 0.f : (pz > fm ? fm : pz);
  int x0 = (int)floorf(px); x0 = x0 < 0 ? 0 : (x0 > D - 2 ? D - 2 : x0);
  int y0 = (int)floorf(py); y0 = y0 < 0 ? 0 : (y0 > D - 2 ? D - 2 : y0);
  int z0 = (int)floorf(pz); z0 = z0 < 0 ? 0 : (z0 > D - 2 ? D - 2 : z0);
  const float fx = px - x0, fy = py - y0, fz = pz - z0;
  const size_t i000 = (size_t)((x0 * D + y0) * D + z0) * C;
  const size_t sx = (size_t)D * D * C, sy = (size_t)D * C, sz = (size_t)C;
  const _Float16* b = src + i000;
  _Float16* o = dst + (size_t)v * C;
  for (int c = 0; c < C; ++c) {
    const float c00 = (float)b[c] * (1.f - fz) + (float)b[c + sz] * fz;
    const float c01 = (float)b[c + sy] * (1.f - fz) + (float)b[c + sy + sz] * fz;
    const float c10 = (float)b[c + sx] * (1.f - fz) + (float)b[c + sx + sz] * fz;
    const float c11 = (float)b[c + sx + sy] * (1.f - fz) + (float)b[c + sx + sy + sz] * fz;
    const float c0 = c00 * (1.f - fy) + c01 * fy;
    const float c1 = c10 * (1.f - fy) + c11 * fy;
    o[c] = (_Float16)(c0 * (1.f - fx) + c1 * fx);
  }
}

// Trilinear align_corners=True resize of f32 [C, iD^3] -> [C, oD^3], * sv.
__global__ void resize_ac_kernel(const float* __restrict__ in,
                                 float* __restrict__ out,
                                 int C, int iD, int oD, float sv) {
  const long long oXYZ = (long long)oD * oD * oD;
  const long long idx = (long long)blockIdx.x * blockDim.x + threadIdx.x;
  if (idx >= (long long)C * oXYZ) return;
  const int c = (int)(idx / oXYZ);
  const long long vv = idx - (long long)c * oXYZ;
  const int z = (int)(vv % oD);
  const int y = (int)((vv / oD) % oD);
  const int x = (int)(vv / ((long long)oD * oD));
  const float r = (oD > 1) ? (float)(iD - 1) / (float)(oD - 1) : 0.f;
  const float px = x * r, py = y * r, pz = z * r;
  int x0 = (int)floorf(px); x0 = x0 < 0 ? 0 : (x0 > iD - 2 ? iD - 2 : x0);
  int y0 = (int)floorf(py); y0 = y0 < 0 ? 0 : (y0 > iD - 2 ? iD - 2 : y0);
  int z0 = (int)floorf(pz); z0 = z0 < 0 ? 0 : (z0 > iD - 2 ? iD - 2 : z0);
  const float fx = px - x0, fy = py - y0, fz = pz - z0;
  const long long iXYZ = (long long)iD * iD * iD;
  const float* b = in + (long long)c * iXYZ + ((long long)(x0 * iD + y0) * iD + z0);
  const int sx = iD * iD, sy = iD;
  const float c00 = b[0] * (1.f - fz) + b[1] * fz;
  const float c01 = b[sy] * (1.f - fz) + b[sy + 1] * fz;
  const float c10 = b[sx] * (1.f - fz) + b[sx + 1] * fz;
  const float c11 = b[sx + sy] * (1.f - fz) + b[sx + sy + 1] * fz;
  const float c0 = c00 * (1.f - fy) + c01 * fy;
  const float c1 = c10 * (1.f - fy) + c11 * fy;
  out[idx] = (c0 * (1.f - fx) + c1 * fx) * sv;
}

// ---------------------------------------------------------------------------
// Host orchestration
// ---------------------------------------------------------------------------
namespace {

struct CW {
  int widx;                 // d_in index of f32 weight tensor
  int cin, cout, kd;
  _Float16* wh;             // permuted f16 weights [Cout, K]
  const float* bias;
};

inline int ceil_div(int a, int b) { return (a + b - 1) / b; }
inline int r8(int c) { return (c + 7) & ~7; }

}  // namespace

extern "C" void kernel_launch(void* const* d_in, const int* in_sizes, int n_in,
                              void* d_out, int out_size, void* d_ws, size_t ws_size,
                              hipStream_t stream) {
  (void)in_sizes; (void)n_in; (void)out_size; (void)ws_size;

  char* arena = (char*)d_ws;
  size_t off = 0;
  auto alloc = [&](size_t bytes) -> void* {
    void* p = arena + off;
    off = (off + bytes + 255) & ~(size_t)255;   // 256-aligned, gives tail slack
    return p;
  };

  const float* x1 = (const float*)d_in[0];
  const float* x2 = (const float*)d_in[1];

  // ---- d_in layout (jax pytree: dict keys sorted, conv leaves b then w) ----
  const int fch[7] = {1, 16, 32, 64, 96, 128, 192};
  CW c11[5], ctx[7], est[6], fea[6], feb[6];
  {
    const int c11ci[5] = {192, 128, 96, 64, 32};
    for (int l = 0; l < 5; ++l)
      c11[l] = {3 + 2 * l, c11ci[l], 32, 1, nullptr, (const float*)d_in[2 + 2 * l]};
    const int cci[7] = {35, 128, 128, 128, 96, 64, 32};
    const int cco[7] = {128, 128, 128, 96, 64, 32, 3};
    for (int i = 0; i < 7; ++i)
      ctx[i] = {13 + 2 * i, cci[i], cco[i], 3, nullptr, (const float*)d_in[12 + 2 * i]};
    const int eci[6] = {764, 128, 256, 224, 160, 96};
    const int eco[6] = {128, 128, 96, 64, 32, 3};
    for (int j = 0; j < 6; ++j)
      est[j] = {27 + 2 * j, eci[j], eco[j], 3, nullptr, (const float*)d_in[26 + 2 * j]};
    for (int i = 0; i < 6; ++i) {
      fea[i] = {39 + 4 * i, fch[i], fch[i + 1], 3, nullptr, (const float*)d_in[38 + 4 * i]};
      feb[i] = {41 + 4 * i, fch[i + 1], fch[i + 1], 3, nullptr, (const float*)d_in[40 + 4 * i]};
    }
  }

  auto prepW = [&](CW& c) {
    const int K3 = c.kd * c.kd * c.kd;
    const int K = K3 * r8(c.cin);
    const int n = c.cout * K;
    c.wh = (_Float16*)alloc((size_t)n * sizeof(_Float16));
    permw_kernel<<<ceil_div(n, 256), 256, 0, stream>>>(
        (const float*)d_in[c.widx], c.wh, c.cout, c.cin, r8(c.cin), K3, K);
  };
  for (int i = 0; i < 5; ++i) prepW(c11[i]);
  for (int i = 0; i < 7; ++i) prepW(ctx[i]);
  for (int i = 0; i < 6; ++i) prepW(est[i]);
  for (int i = 0; i < 6; ++i) { prepW(fea[i]); prepW(feb[i]); }

  auto conv = [&](const CW& c, const _Float16* in, int inStrideC,
                  _Float16* o16, int outStrideC, float* o32,
                  int dOut, int dIn, int strideV, int dilV, int reluV) {
    ConvArgs A;
    A.in = in; A.wt = c.wh; A.bias = c.bias; A.out16 = o16; A.out32 = o32;
    A.inStrideC = inStrideC; A.outStrideC = outStrideC;
    A.Cin = c.cin; A.CinP = r8(c.cin); A.Cout = c.cout;
    A.K = c.kd * c.kd * c.kd * A.CinP;
    A.M = dOut * dOut * dOut;
    A.d = dOut; A.zb = __builtin_ctz((unsigned)dOut);
    A.inD = dIn; A.stride = strideV; A.dil = dilV;
    A.pad = (c.kd - 1) * dilV / 2; A.relu = reluV;
    dim3 grid(ceil_div(A.M, 64), ceil_div(c.cout, 32));
    if (c.kd == 1) conv3d_wmma<1><<<grid, 256, 0, stream>>>(A);
    else           conv3d_wmma<3><<<grid, 256, 0, stream>>>(A);
  };

  // ---- inputs -> f16 channel-last (stride 8) ----
  const int S3 = 128 * 128 * 128;
  _Float16* x1h = (_Float16*)alloc((size_t)S3 * 8 * 2);
  _Float16* x2h = (_Float16*)alloc((size_t)S3 * 8 * 2);
  cast_in_kernel<<<ceil_div(S3, 256), 256, 0, stream>>>(x1, x1h, S3);
  cast_in_kernel<<<ceil_div(S3, 256), 256, 0, stream>>>(x2, x2h, S3);

  // ---- feature pyramids (f16 channel-last, stride = C) ----
  _Float16* f1[6]; _Float16* f2[6];
  _Float16* tmpFeat = (_Float16*)alloc((size_t)16 * 64 * 64 * 64 * 2);
  for (int which = 0; which < 2; ++which) {
    const _Float16* cur = which ? x2h : x1h;
    int curStride = 8;
    _Float16** fo = which ? f2 : f1;
    int dprev = 128;
    for (int i = 0; i < 6; ++i) {
      const int dOut = dprev / 2;
      const size_t n = (size_t)fch[i + 1] * dOut * dOut * dOut;
      conv(fea[i], cur, curStride, tmpFeat, fch[i + 1], nullptr, dOut, dprev, 2, 1, 1);
      fo[i] = (_Float16*)alloc(n * 2);
      conv(feb[i], tmpFeat, fch[i + 1], fo[i], fch[i + 1], nullptr, dOut, dOut, 1, 1, 1);
      cur = fo[i]; curStride = fch[i + 1]; dprev = dOut;
    }
  }

  // ---- shared big buffers (channel-last, sized for finest level 32^3) ----
  const int XYZmax = 32 * 32 * 32;
  const int CONCAT_S = 768;   // 729 + 32 + 3 -> padded to 768
  const int EST_S = 448;      // 128+128+96+64+32
  const int CTXIN_S = 40;     // 35 -> 40
  _Float16* concatB = (_Float16*)alloc((size_t)CONCAT_S * XYZmax * 2);
  _Float16* estB    = (_Float16*)alloc((size_t)EST_S * XYZmax * 2);
  _Float16* ctxinB  = (_Float16*)alloc((size_t)CTXIN_S * XYZmax * 2);
  _Float16* ctxP[2] = {(_Float16*)alloc((size_t)128 * XYZmax * 2),
                       (_Float16*)alloc((size_t)128 * XYZmax * 2)};
  _Float16* warpB   = (_Float16*)alloc((size_t)32 * XYZmax * 2);  // max C*XYZ
  float* fresB  = (float*)alloc((size_t)3 * XYZmax * 4);
  float* ffineB = (float*)alloc((size_t)3 * XYZmax * 4);
  float* flowL[5];
  for (int l = 0; l < 5; ++l) {
    const int d = 2 << l;
    flowL[l] = (float*)alloc((size_t)3 * d * d * d * 4);
  }

  // ---- coarse-to-fine flow estimation ----
  const int dils[7] = {1, 2, 4, 8, 16, 1, 1};
  for (int l = 0; l < 5; ++l) {
    const int d = 2 << l;
    const int XYZ = d * d * d;
    const int fi = 5 - l;
    const int C = fch[fi + 1];
    const _Float16* X1f = f1[fi];
    const _Float16* X2f = f2[fi];
    const _Float16* warpSrc;
    const int gv = ceil_div(XYZ, 256);
    const int g3 = ceil_div(3 * XYZ, 256);

    if (l == 0) {
      fill_f32_kernel<<<g3, 256, 0, stream>>>(flowL[0], 0.f, 3 * XYZ);
      warpSrc = X2f;
    } else {
      const int dp = d / 2;
      resize_ac_kernel<<<g3, 256, 0, stream>>>(flowL[l - 1], flowL[l], 3, dp, d, 2.0f);
      warp_kernel<<<gv, 256, 0, stream>>>(X2f, flowL[l], warpB, C, d);
      warpSrc = warpB;
    }

    // correlation -> concat channels [0,729), lrelu fused
    dim3 gcorr(gv, 729);
    corr_kernel<<<gcorr, 256, 0, stream>>>(X1f, warpSrc, concatB, C, C, d, CONCAT_S);
    // 1x1 conv of x1 features -> concat channels [729,761)
    conv(c11[l], X1f, C, concatB + 729, CONCAT_S, nullptr, d, d, 1, 1, 1);
    // flow -> concat channels [761,764)
    cast_flowcl_kernel<<<g3, 256, 0, stream>>>(flowL[l], concatB + 761, XYZ, CONCAT_S);

    // estimator; channel-last slices make DenseNet concats contiguous
    conv(est[0], concatB, CONCAT_S, estB, EST_S, nullptr, d, d, 1, 1, 1);        // ch 0..127
    conv(est[1], estB, EST_S, estB + 128, EST_S, nullptr, d, d, 1, 1, 1);        // 128..255
    conv(est[2], estB, EST_S, estB + 256, EST_S, nullptr, d, d, 1, 1, 1);        // 256..351
    conv(est[3], estB + 128, EST_S, estB + 352, EST_S, nullptr, d, d, 1, 1, 1);  // 352..415
    conv(est[4], estB + 256, EST_S, estB + 416, EST_S, nullptr, d, d, 1, 1, 1);  // 416..447
    conv(est[5], estB + 352, EST_S, nullptr, 0, fresB, d, d, 1, 1, 0);           // flow res
    add_f32_kernel<<<g3, 256, 0, stream>>>(flowL[l], fresB, 3 * XYZ);

    // context net on concat(xi, flow)
    copy_chcl_kernel<<<ceil_div(XYZ * 32, 256), 256, 0, stream>>>(
        estB + 416, ctxinB, XYZ, 32, EST_S, CTXIN_S);
    cast_flowcl_kernel<<<g3, 256, 0, stream>>>(flowL[l], ctxinB + 32, XYZ, CTXIN_S);
    const _Float16* ci = ctxinB;
    int cis = CTXIN_S;
    for (int i = 0; i < 6; ++i) {
      conv(ctx[i], ci, cis, ctxP[i & 1], ctx[i].cout, nullptr, d, d, 1, dils[i], 1);
      ci = ctxP[i & 1]; cis = ctx[i].cout;
    }
    conv(ctx[6], ci, cis, nullptr, 0, ffineB, d, d, 1, dils[6], 0);
    add_f32_kernel<<<g3, 256, 0, stream>>>(flowL[l], ffineB, 3 * XYZ);
  }

  // ---- outputs: flows[::-1], each resized x4 with values * 4 ----
  float* outp = (float*)d_out;
  size_t ooff = 0;
  for (int l = 4; l >= 0; --l) {
    const int d = 2 << l;
    const int od = d * 4;
    const size_t n = (size_t)3 * od * od * od;
    resize_ac_kernel<<<ceil_div((int)n, 256), 256, 0, stream>>>(
        flowL[l], outp + ooff, 3, d, od, 4.0f);
    ooff += n;
  }
}
